// CrossAttention_12936441496003
// MI455X (gfx1250) — compile-verified, matrix operand order checked by reading, not run on hardware
//
#include <hip/hip_runtime.h>

// ---------------------------------------------------------------------------
// CDNA5 / gfx1250 cross-attention: bf16 WMMA everywhere, f32 accumulate.
// ---------------------------------------------------------------------------

typedef __bf16 bf16;
typedef __attribute__((ext_vector_type(16))) __bf16 v16bf;
typedef __attribute__((ext_vector_type(8)))  float  v8f;

union FragBF {
    v16bf v;
    bf16  e[16];
    uint4 q[2];
};
union AccF {
    v8f   v;
    float f[8];
};

// Load a 16x32 bf16 A-fragment (or mirrored B-fragment from a [n][k]-stored
// tile). Per ISA 7.12.2: lanes 0-15 hold row m, k = 0..7 (VGPR0-3) and
// k = 16..23 (VGPR4-7); lanes 16-31 hold row m, k = 8..15 and 24..31.
// => per lane: two contiguous 16-byte chunks at k0 = half*8 and k0+16.
__device__ inline v16bf load_frag(const bf16* base, int stride, int lane) {
    const int half = (lane >> 4) & 1;
    const int m    = lane & 15;
    const bf16* p  = base + m * stride + half * 8;
    FragBF f;
    f.q[0] = *(const uint4*)(p);
    f.q[1] = *(const uint4*)(p + 16);
    return f.v;
}

__device__ inline v8f wmma_bf16(v16bf a, v16bf b, v8f c) {
    return __builtin_amdgcn_wmma_f32_16x16x32_bf16(
        /*neg_a=*/false, a, /*neg_b=*/false, b,
        /*c_mod=*/(short)0, c, /*reuse_a=*/false, /*reuse_b=*/false);
}

// ---------------------------------------------------------------------------
// fp32 -> bf16 conversion (with optional scale folding, used to fold the
// softmax SCALE=1/8 into Wq once).
// ---------------------------------------------------------------------------
__global__ void cvt_f32_bf16(const float* __restrict__ src, bf16* __restrict__ dst,
                             int n, float scale) {
    int stride = gridDim.x * blockDim.x * 4;
    for (int i = (blockIdx.x * blockDim.x + threadIdx.x) * 4; i < n; i += stride) {
        float4 v = *(const float4*)(src + i);
        dst[i + 0] = (bf16)(v.x * scale);
        dst[i + 1] = (bf16)(v.y * scale);
        dst[i + 2] = (bf16)(v.z * scale);
        dst[i + 3] = (bf16)(v.w * scale);
    }
}

// ---------------------------------------------------------------------------
// Tiled bf16 GEMM:  C[M,N] = A[M,K] @ B[K,N]   (A,B bf16 row-major)
// Block: 256 threads = 8 waves; block tile 128x128; wave tile 64x32;
// K-step 32. B tile stored transposed in LDS so fragment reads are
// contiguous per lane. F32OUT adds bias and writes fp32, else bf16.
// Requires M%128==0, N%128==0, K%32==0 (true for all shapes here).
// ---------------------------------------------------------------------------
#define GBM 128
#define GBN 128
#define GBK 32
#define GLD 40   // padded LDS row stride (multiple of 8 for 16B alignment)

template <bool F32OUT>
__global__ __launch_bounds__(256) void gemm_bf16(
    const bf16* __restrict__ A, const bf16* __restrict__ B,
    void* __restrict__ Cout, const float* __restrict__ bias,
    int M, int N, int K) {
    __shared__ bf16 As[GBM * GLD];
    __shared__ bf16 Bs[GBN * GLD];   // transposed: [n][k]

    const int tid  = threadIdx.x;
    const int lane = tid & 31;
    const int wave = tid >> 5;       // 0..7
    const int wm   = wave >> 2;      // 0..1  -> 64 rows
    const int wn   = wave & 3;       // 0..3  -> 32 cols
    const int rowBase = blockIdx.y * GBM;
    const int colBase = blockIdx.x * GBN;

    AccF acc[4][2];
#pragma unroll
    for (int i = 0; i < 4; ++i)
#pragma unroll
        for (int j = 0; j < 2; ++j) acc[i][j].v = (v8f){};

    for (int k0 = 0; k0 < K; k0 += GBK) {
        // A tile: 128x32 -> 512 chunks of 8 bf16
#pragma unroll
        for (int i = 0; i < 2; ++i) {
            int c  = tid + i * 256;
            int r  = c >> 2;
            int kc = (c & 3) * 8;
            uint4 d = *(const uint4*)(A + (size_t)(rowBase + r) * K + k0 + kc);
            *(uint4*)(&As[r * GLD + kc]) = d;
        }
        // B tile: 32x128, scattered into transposed LDS layout
#pragma unroll
        for (int i = 0; i < 2; ++i) {
            int c  = tid + i * 256;
            int kk = c >> 4;          // 0..31
            int nc = (c & 15) * 8;    // 0..120
            union { uint4 q; bf16 e[8]; } d;
            d.q = *(const uint4*)(B + (size_t)(k0 + kk) * N + colBase + nc);
#pragma unroll
            for (int j = 0; j < 8; ++j) Bs[(nc + j) * GLD + kk] = d.e[j];
        }
        __syncthreads();

        v16bf af[4], bfr[2];
#pragma unroll
        for (int i = 0; i < 4; ++i)
            af[i] = load_frag(&As[(wm * 64 + i * 16) * GLD], GLD, lane);
#pragma unroll
        for (int j = 0; j < 2; ++j)
            bfr[j] = load_frag(&Bs[(wn * 32 + j * 16) * GLD], GLD, lane);
#pragma unroll
        for (int i = 0; i < 4; ++i)
#pragma unroll
            for (int j = 0; j < 2; ++j)
                acc[i][j].v = wmma_bf16(af[i], bfr[j], acc[i][j].v);
        __syncthreads();
    }

    // Epilogue. C/D layout: VGPR r -> row r (lanes 0-15) / r+8 (lanes 16-31).
    const int half = (lane >> 4) & 1;
    const int nl   = lane & 15;
#pragma unroll
    for (int i = 0; i < 4; ++i) {
#pragma unroll
        for (int j = 0; j < 2; ++j) {
            int col = colBase + wn * 32 + j * 16 + nl;
            float bv = F32OUT ? bias[col] : 0.0f;
#pragma unroll
            for (int r = 0; r < 8; ++r) {
                int row = rowBase + wm * 64 + i * 16 + r + 8 * half;
                if (F32OUT)
                    ((float*)Cout)[(size_t)row * N + col] = acc[i][j].f[r] + bv;
                else
                    ((bf16*)Cout)[(size_t)row * N + col] = (bf16)acc[i][j].f[r];
            }
        }
    }
}

// ---------------------------------------------------------------------------
// Attention: one workgroup per (b, h, 64-query block). Full K (512x64) and
// V^T (64x512) for the head live in LDS (dynamic, ~148 KB of the 320 KB WGP
// budget). 4 waves x 16 query rows; online softmax; P bounced through a
// per-wave LDS staging tile to convert C/D layout -> A-fragment layout.
// SCALE is pre-folded into Wq, so scores need no extra scaling here.
// ---------------------------------------------------------------------------
#define KS_STRIDE 80    // 64 + pad, multiple of 8
#define VT_STRIDE 520   // 512 + pad, multiple of 8
#define PS_STRIDE 40    // 32 + pad, multiple of 8
#define ATTN_SMEM ((size_t)512 * KS_STRIDE * 2 + (size_t)64 * VT_STRIDE * 2 + (size_t)4 * 16 * PS_STRIDE * 2)

__global__ __launch_bounds__(128) void attn_kernel(
    const bf16* __restrict__ Q, const bf16* __restrict__ Kc,
    const bf16* __restrict__ Vc, bf16* __restrict__ O) {
    extern __shared__ char smem[];
    bf16* Ks = (bf16*)smem;                                      // [512][80]
    bf16* Vt = (bf16*)(smem + (size_t)512 * KS_STRIDE * 2);      // [64][520]
    bf16* Ps = (bf16*)(smem + (size_t)512 * KS_STRIDE * 2
                            + (size_t)64 * VT_STRIDE * 2);       // [4][16][40]

    const int tid  = threadIdx.x;
    const int lane = tid & 31;
    const int wave = tid >> 5;
    const int qb   = blockIdx.x;
    const int h    = blockIdx.y;
    const int b    = blockIdx.z;
    const int HD = 1024, N = 4096, M = 512;

    // Cooperative load of K and V^T for this (b, h).
#pragma unroll 4
    for (int i = 0; i < 32; ++i) {
        int c  = tid + i * 128;          // 0..4095
        int r  = c >> 3;                 // key row 0..511
        int dc = (c & 7) * 8;            // d offset 0..56
        const size_t g = (size_t)(b * M + r) * HD + h * 64 + dc;
        uint4 kd = *(const uint4*)(Kc + g);
        *(uint4*)(&Ks[r * KS_STRIDE + dc]) = kd;
        union { uint4 q; bf16 e[8]; } vd;
        vd.q = *(const uint4*)(Vc + g);
#pragma unroll
        for (int j = 0; j < 8; ++j) Vt[(dc + j) * VT_STRIDE + r] = vd.e[j];
    }
    __syncthreads();

    const int  qrow  = qb * 64 + wave * 16;
    const bf16* qbase = Q + (size_t)(b * N + qrow) * HD + h * 64;
    v16bf aq0 = load_frag(qbase,      HD, lane);   // d = 0..31
    v16bf aq1 = load_frag(qbase + 32, HD, lane);   // d = 32..63

    AccF o[4];
#pragma unroll
    for (int j = 0; j < 4; ++j) o[j].v = (v8f){};
    float rmax[8], rsum[8];
#pragma unroll
    for (int r = 0; r < 8; ++r) { rmax[r] = -1e30f; rsum[r] = 0.0f; }

    bf16*     psw  = Ps + wave * 16 * PS_STRIDE;
    const int half = (lane >> 4) & 1;
    const int nl   = lane & 15;

    for (int kb = 0; kb < M; kb += 32) {
        v8f z = (v8f){};
        AccF s0, s1;   // scores vs keys kb..kb+15 and kb+16..kb+31
        s0.v = wmma_bf16(aq0, load_frag(&Ks[(kb +  0) * KS_STRIDE +  0], KS_STRIDE, lane), z);
        s0.v = wmma_bf16(aq1, load_frag(&Ks[(kb +  0) * KS_STRIDE + 32], KS_STRIDE, lane), s0.v);
        s1.v = wmma_bf16(aq0, load_frag(&Ks[(kb + 16) * KS_STRIDE +  0], KS_STRIDE, lane), z);
        s1.v = wmma_bf16(aq1, load_frag(&Ks[(kb + 16) * KS_STRIDE + 32], KS_STRIDE, lane), s1.v);

        // Online softmax update over these 32 keys; row reductions over the
        // 16-lane half that owns the row (wave32: xor masks 1,2,4,8).
#pragma unroll
        for (int r = 0; r < 8; ++r) {
            float t = fmaxf(s0.f[r], s1.f[r]);
            t = fmaxf(t, __shfl_xor(t, 1, 32));
            t = fmaxf(t, __shfl_xor(t, 2, 32));
            t = fmaxf(t, __shfl_xor(t, 4, 32));
            t = fmaxf(t, __shfl_xor(t, 8, 32));
            float nm   = fmaxf(rmax[r], t);
            float corr = __expf(rmax[r] - nm);
            float p0   = __expf(s0.f[r] - nm);
            float p1   = __expf(s1.f[r] - nm);
            float ps   = p0 + p1;
            ps += __shfl_xor(ps, 1, 32);
            ps += __shfl_xor(ps, 2, 32);
            ps += __shfl_xor(ps, 4, 32);
            ps += __shfl_xor(ps, 8, 32);
            rsum[r] = rsum[r] * corr + ps;
            rmax[r] = nm;
#pragma unroll
            for (int j = 0; j < 4; ++j) o[j].f[r] *= corr;
            // Stage P (bf16) in A-consumable LDS layout: row = r+8*half.
            psw[(r + 8 * half) * PS_STRIDE + nl]      = (bf16)p0;
            psw[(r + 8 * half) * PS_STRIDE + 16 + nl] = (bf16)p1;
        }
        // Per-wave staging: LDS ops from the same wave are in-order (DScnt),
        // so the fragment load below sees the stores above.
        v16bf pa = load_frag(psw, PS_STRIDE, lane);
#pragma unroll
        for (int j = 0; j < 4; ++j)
            o[j].v = wmma_bf16(pa, load_frag(&Vt[(j * 16) * VT_STRIDE + kb], VT_STRIDE, lane), o[j].v);
    }

    // Normalize and write O in [B, N, H*D] bf16.
#pragma unroll
    for (int r = 0; r < 8; ++r) {
        float inv = 1.0f / rsum[r];
        int row = qrow + r + 8 * half;
#pragma unroll
        for (int j = 0; j < 4; ++j)
            O[(size_t)(b * N + row) * HD + h * 64 + j * 16 + nl] =
                (bf16)(o[j].f[r] * inv);
    }
}

// ---------------------------------------------------------------------------
// Host launcher
// ---------------------------------------------------------------------------
extern "C" void kernel_launch(void* const* d_in, const int* in_sizes, int n_in,
                              void* d_out, int out_size, void* d_ws, size_t ws_size,
                              hipStream_t stream) {
    (void)in_sizes; (void)n_in; (void)out_size; (void)ws_size;
    const float* x   = (const float*)d_in[0];
    const float* ctx = (const float*)d_in[1];
    const float* Wq  = (const float*)d_in[2];
    const float* Wk  = (const float*)d_in[3];
    const float* Wv  = (const float*)d_in[4];
    const float* Wo  = (const float*)d_in[5];
    const float* bo  = (const float*)d_in[6];
    float* out = (float*)d_out;

    const int B = 4, N = 4096, C = 1024, M = 512, Cc = 768, HD = 1024;
    const float SCALE = 0.125f;   // 64^-0.5, folded into Wq

    char* ws = (char*)d_ws;
    size_t off = 0;
    auto carve = [&](size_t bytes) -> bf16* {
        bf16* p = (bf16*)(ws + off);
        off += (bytes + 255) & ~(size_t)255;
        return p;
    };
    bf16* xb  = carve((size_t)B * N * C * 2);
    bf16* cb  = carve((size_t)B * M * Cc * 2);
    bf16* wqb = carve((size_t)C * HD * 2);
    bf16* wkb = carve((size_t)Cc * HD * 2);
    bf16* wvb = carve((size_t)Cc * HD * 2);
    bf16* wob = carve((size_t)HD * C * 2);
    bf16* Qb  = carve((size_t)B * N * HD * 2);
    bf16* Kb  = carve((size_t)B * M * HD * 2);
    bf16* Vb  = carve((size_t)B * M * HD * 2);
    bf16* Ob  = carve((size_t)B * N * HD * 2);

    auto cvt = [&](const float* s, bf16* d, int n, float sc) {
        int blocks = (n / 4 + 255) / 256;
        if (blocks > 4096) blocks = 4096;
        cvt_f32_bf16<<<dim3(blocks), dim3(256), 0, stream>>>(s, d, n, sc);
    };
    cvt(x,   xb,  B * N * C,  1.0f);
    cvt(ctx, cb,  B * M * Cc, 1.0f);
    cvt(Wq,  wqb, C * HD,     SCALE);
    cvt(Wk,  wkb, Cc * HD,    1.0f);
    cvt(Wv,  wvb, Cc * HD,    1.0f);
    cvt(Wo,  wob, HD * C,     1.0f);

    // Projections (bf16 out): Q = x@Wq*SCALE, K = ctx@Wk, V = ctx@Wv
    gemm_bf16<false><<<dim3(HD / GBN, (B * N) / GBM), 256, 0, stream>>>(
        xb, wqb, Qb, nullptr, B * N, HD, C);
    gemm_bf16<false><<<dim3(HD / GBN, (B * M) / GBM), 256, 0, stream>>>(
        cb, wkb, Kb, nullptr, B * M, HD, Cc);
    gemm_bf16<false><<<dim3(HD / GBN, (B * M) / GBM), 256, 0, stream>>>(
        cb, wvb, Vb, nullptr, B * M, HD, Cc);

    // Attention: grid = (query blocks, heads, batch)
    attn_kernel<<<dim3(N / 64, 16, B), 128, ATTN_SMEM, stream>>>(Qb, Kb, Vb, Ob);

    // Output projection (fp32 out + bias)
    gemm_bf16<true><<<dim3(C / GBN, (B * N) / GBM), 256, 0, stream>>>(
        Ob, wob, out, bo, B * N, C, HD);
}
